// ROIAlign3d_65051574665802
// MI455X (gfx1250) — matrix operand order repaired and checked by reading, not compiled, not run on hardware
//
#include <hip/hip_runtime.h>

// ROIAlign3d as per-ROI GEMM:  out[M=C*L, N=49] = feat[M, K=256] x W^T[K, N]
// f16 WMMA (16x16x32) with f32 accumulate -> matrix work ~6.5us, at/below the
// ~8.4us HBM store floor (196MB @ 23.3TB/s). Round 3: each wave owns TWO
// 16-row M-tiles so every B fragment fetched from LDS feeds two independent
// WMMAs -> half the LDS traffic and half the s_wait_dscnt stalls per matrix op.

typedef __attribute__((ext_vector_type(8)))  float    v8f;
typedef __attribute__((ext_vector_type(8)))  _Float16 v8h;
typedef __attribute__((ext_vector_type(16))) _Float16 v16h;

#define PH_    7
#define PW_    7
#define SR_    2
#define SCALE_ 0.0625f
#define C_     1024
#define L_     32
#define H_     16
#define W_     16
#define KDIM   256     // H*W
#define KS2    264     // f16 LDS row stride: 528B rows -> dword stride 132 == 4 (mod 64), conflict-free
#define NPAD   64      // 49 bins padded to 4 x 16 N-tiles
#define MROWS  32768   // C*L
#define NBINS  49
#define WAVES  8       // 256-thread block = 8 wave32
#define MT     2       // M-tiles per wave

__global__ __launch_bounds__(256) void roialign3d_wmma_f16_kernel(
    const float* __restrict__ input,   // [B, C, L, H, W] fp32
    const float* __restrict__ rois,    // [N, 5] fp32
    float* __restrict__ out)           // [N, C, L, PH, PW] fp32
{
  __shared__ _Float16 Wh[NPAD * KS2];  // W^T in f16, [N=64][K stride 264], ~33 KB

  const int tid    = threadIdx.x;
  const int roi    = blockIdx.x;
  const int lane   = tid & 31;
  const int wave   = tid >> 5;
  const int laneLo = lane & 15;        // M (A) / N (B,D) position within a tile
  const int laneHi = lane >> 4;        // A: +8 K-run select; B: +8 K-run select; D: +8 M rows

  // ---- zero the f16 weight matrix ----
  {
    unsigned int* wz = (unsigned int*)Wh;
    for (int i = tid; i < (NPAD * KS2) / 2; i += 256) wz[i] = 0u;
  }
  __syncthreads();

  // ---- build per-ROI bilinear weights (one thread per output bin) ----
  const float* R = rois + roi * 5;
  if (tid < NBINS) {
    const int ph = tid / PW_, pw = tid % PW_;
    const float x1 = R[1] * SCALE_, y1 = R[2] * SCALE_;
    const float x2 = R[3] * SCALE_, y2 = R[4] * SCALE_;
    const float rw = fmaxf(x2 - x1, 1.0f), rh = fmaxf(y2 - y1, 1.0f);
    const float bw = rw / (float)PW_,  bh = rh / (float)PH_;
    _Float16* wrow = &Wh[tid * KS2];   // this thread owns the whole row -> race-free RMW
    for (int sy = 0; sy < SR_; ++sy) {
      for (int sx = 0; sx < SR_; ++sx) {
        const float yc = y1 + (float)ph * bh + ((float)sy + 0.5f) * (bh / (float)SR_);
        const float xc = x1 + (float)pw * bw + ((float)sx + 0.5f) * (bw / (float)SR_);
        if (yc < -1.0f || yc > (float)H_ || xc < -1.0f || xc > (float)W_) continue;
        const float cy = fminf(fmaxf(yc, 0.0f), (float)(H_ - 1));
        const float cx = fminf(fmaxf(xc, 0.0f), (float)(W_ - 1));
        int y0 = (int)floorf(cy); if (y0 > H_ - 1) y0 = H_ - 1;
        int x0 = (int)floorf(cx); if (x0 > W_ - 1) x0 = W_ - 1;
        const int y1i = (y0 + 1 > H_ - 1) ? H_ - 1 : y0 + 1;
        const int x1i = (x0 + 1 > W_ - 1) ? W_ - 1 : x0 + 1;
        const float ly = cy - (float)y0, hy = 1.0f - ly;
        const float lx = cx - (float)x0, hx = 1.0f - lx;
        const float q = 1.0f / (float)(SR_ * SR_);
        const int i00 = y0  * W_ + x0,  i01 = y0  * W_ + x1i;
        const int i10 = y1i * W_ + x0,  i11 = y1i * W_ + x1i;
        wrow[i00] = (_Float16)((float)wrow[i00] + q * hy * hx);
        wrow[i01] = (_Float16)((float)wrow[i01] + q * hy * lx);
        wrow[i10] = (_Float16)((float)wrow[i10] + q * ly * hx);
        wrow[i11] = (_Float16)((float)wrow[i11] + q * ly * lx);
      }
    }
  }
  __syncthreads();

  // ---- GEMM: each wave owns two 16-row M-tiles, 4 N-tiles, K = 256 ----
  const int bidx = (int)R[0];
  const float* feat = input + (size_t)bidx * (size_t)C_ * L_ * KDIM;
  const int m0 = (blockIdx.y * WAVES + wave) * (16 * MT);

  // f16 A (16x32) layout: lane holds M = m0+laneLo, K runs {b..b+7, b+16..b+23}, b = 8*laneHi.
  const float* aRow0 = feat + (size_t)(m0 + laneLo) * KDIM + 8 * laneHi;
  const float* aRow1 = aRow0 + (size_t)16 * KDIM;
  // f16 B (32x16) layout: lane holds N = n0+laneLo, K runs {b..b+7, b+16..b+23}, b = 8*laneHi.
  const _Float16* bRow = &Wh[laneLo * KS2 + 8 * laneHi];

  v8f acc0[4] = {};   // M-tile 0, N-tiles 0..3
  v8f acc1[4] = {};   // M-tile 1, N-tiles 0..3

  #pragma unroll
  for (int k = 0; k < KDIM; k += 32) {
    // A fragments: 32B f32 loads, cvt to f16 in-register (co-executes with WMMA)
    #define A_FRAG(P)                                                          \
      __builtin_shufflevector(                                                 \
          __builtin_convertvector(*(const v8f*)((P) + k),      v8h),           \
          __builtin_convertvector(*(const v8f*)((P) + k + 16), v8h),           \
          0, 1, 2, 3, 4, 5, 6, 7, 8, 9, 10, 11, 12, 13, 14, 15)
    const v16h a0 = A_FRAG(aRow0);
    const v16h a1 = A_FRAG(aRow1);
    #undef A_FRAG

    #pragma unroll
    for (int t = 0; t < 4; ++t) {
      // one B fragment feeds two independent WMMAs
      const v16h b = __builtin_shufflevector(
          *(const v8h*)(bRow + t * 16 * KS2 + k),
          *(const v8h*)(bRow + t * 16 * KS2 + k + 16),
          0, 1, 2, 3, 4, 5, 6, 7, 8, 9, 10, 11, 12, 13, 14, 15);
      acc0[t] = __builtin_amdgcn_wmma_f32_16x16x32_f16(false, a0, false, b, (short)0, acc0[t], false, false);
      acc1[t] = __builtin_amdgcn_wmma_f32_16x16x32_f16(false, a1, false, b, (short)0, acc1[t], false, false);
    }
  }

  // ---- D writeback: lane = N column, VGPR r = M row (+8 for laneHi) ----
  #pragma unroll
  for (int mt = 0; mt < MT; ++mt) {
    const size_t obase =
        ((size_t)roi * MROWS + (size_t)(m0 + 16 * mt + 8 * laneHi)) * NBINS;
    const v8f* accs = mt ? acc1 : acc0;
    #pragma unroll
    for (int t = 0; t < 4; ++t) {
      const int col = t * 16 + laneLo;
      if (col < NBINS) {
        #pragma unroll
        for (int r = 0; r < 8; ++r)
          out[obase + (size_t)r * NBINS + (size_t)col] = accs[t][r];
      }
    }
  }
}

extern "C" void kernel_launch(void* const* d_in, const int* in_sizes, int n_in,
                              void* d_out, int out_size, void* d_ws, size_t ws_size,
                              hipStream_t stream) {
  const float* input = (const float*)d_in[0];
  const float* rois  = (const float*)d_in[1];
  float* out = (float*)d_out;

  const int n_rois  = in_sizes[1] / 5;             // 32
  const int mchunks = MROWS / (WAVES * 16 * MT);   // 32768 / 256 = 128
  dim3 grid(n_rois, mchunks, 1);
  roialign3d_wmma_f16_kernel<<<grid, 256, 0, stream>>>(input, rois, out);
}
